// MessagePassingProcessorWraper_57011395887383
// MI455X (gfx1250) — compile-verified, matrix operand order checked by reading, not run on hardware
//
#include <hip/hip_runtime.h>

// ---------------------------------------------------------------------------
// MessagePassing GNN block for MI455X (gfx1250), bf16 WMMA + fp32 accum.
//   N_NODES=50000, N_EDGES=512000, HID=128, EDGE_DIM=3, 2 layers.
// One wave32 owns a 16-row tile and computes its full 16x128 MLP output with
// v_wmma_f32_16x16x32_bf16 (8 N-tiles x K/32 steps), LDS-staged between the
// two GEMMs, LayerNorm done in fp32 with cross-lane shuffles.
// Conversions use native __bf16 casts so clang emits v_cvt_*bf16* instead of
// 3-op integer rounding sequences.
// ---------------------------------------------------------------------------

#define N_NODES 50000
#define N_EDGES 512000
#define HID     128
#define LN_EPS  1e-5f

#define STRH 136   // bf16 row stride in LDS staging (padded: banks step by 4)
#define STRF 132   // f32 row stride in LDS LN buffer

typedef __attribute__((ext_vector_type(16))) __bf16 v16bf;
typedef __attribute__((ext_vector_type(8)))  float  v8f;

static __device__ __forceinline__ __bf16 f2bf(float f) { return (__bf16)f; }
static __device__ __forceinline__ float  bf2f(__bf16 b) { return (float)b; }
static __device__ __forceinline__ float silu(float x) {
  return x / (1.0f + __expf(-x));
}

static __device__ __forceinline__ v8f wmma_bf16(v16bf a, v16bf b, v8f c) {
  // D = A(16x32) * B(32x16) + C(16x16), fp32 accumulate
  return __builtin_amdgcn_wmma_f32_16x16x32_bf16(false, a, false, b,
                                                 (short)0, c, false, false);
}

// Packed weight fragment: [ktile][ntile=8][lane=32][h=16] bf16, 32B per lane.
static __device__ __forceinline__ v16bf loadB(const __bf16* __restrict__ Wp,
                                              int kt, int t, int lane) {
  return *(const v16bf*)(Wp + (((size_t)kt * 8 + t) * 512) + lane * 16);
}

// A fragment (16x32 bf16): lane<16 holds row lh, K {k..k+7, k+16..k+23};
// lane>=16 same row, K {k+8..k+15, k+24..k+31}.
static __device__ __forceinline__ void fill_a_f32(v16bf& a, const float* p,
                                                  int kBase, int hb) {
  const float* s0 = p + kBase + hb * 8;
  const float* s1 = s0 + 16;
#pragma unroll
  for (int i = 0; i < 8; ++i) { a[i] = f2bf(s0[i]); a[i + 8] = f2bf(s1[i]); }
}
static __device__ __forceinline__ void fill_a_bf16(v16bf& a, const __bf16* p,
                                                   int kBase, int hb) {
  const __bf16* s0 = p + kBase + hb * 8;
  const __bf16* s1 = s0 + 16;
#pragma unroll
  for (int i = 0; i < 8; ++i) { a[i] = s0[i]; a[i + 8] = s1[i]; }
}

// ---------------------------------------------------------------------------
// Shared tail: h2 = silu(hbuf(16x128 bf16) @ W2 + b2); LayerNorm(g,beta).
// Leaves final LN output in fbuf (16 x STRF f32). Must be called by all
// threads of the block (contains __syncthreads).
// ---------------------------------------------------------------------------
static __device__ __forceinline__ void gemm2_silu_ln(
    const __bf16* __restrict__ hbuf, float* __restrict__ fbuf,
    const __bf16* __restrict__ W2p, const float* __restrict__ b2,
    const float* __restrict__ gm, const float* __restrict__ bt,
    int lh, int hb, int lane) {
  v8f acc[8] = {};
  const __bf16* arow = hbuf + lh * STRH;
#pragma unroll
  for (int ks = 0; ks < 4; ++ks) {
    v16bf a;
#pragma unroll
    for (int i = 0; i < 8; ++i) {
      a[i]     = arow[ks * 32 + hb * 8 + i];
      a[i + 8] = arow[ks * 32 + 16 + hb * 8 + i];
    }
#pragma unroll
    for (int t = 0; t < 8; ++t)
      acc[t] = wmma_bf16(a, loadB(W2p, ks, t, lane), acc[t]);
  }
  // bias + silu -> fp32 LDS (C layout: row g+hb*8, col t*16+lh)
#pragma unroll
  for (int t = 0; t < 8; ++t) {
    float bias = b2[t * 16 + lh];
#pragma unroll
    for (int g = 0; g < 8; ++g)
      fbuf[(g + hb * 8) * STRF + t * 16 + lh] = silu(acc[t][g] + bias);
  }
  __syncthreads();
  // LayerNorm: 2 lanes per row, each reduces 64 values, combine via xor-16.
  float s = 0.f, s2 = 0.f;
  float* row = fbuf + lh * STRF + hb * 64;
  for (int i = 0; i < 64; ++i) { float v = row[i]; s += v; s2 += v * v; }
  s  += __shfl_xor(s, 16, 32);
  s2 += __shfl_xor(s2, 16, 32);
  float mu  = s * (1.f / 128.f);
  float var = s2 * (1.f / 128.f) - mu * mu;
  float rs  = rsqrtf((var > 0.f ? var : 0.f) + LN_EPS);
  for (int i = 0; i < 64; ++i) {
    int col = hb * 64 + i;
    row[i] = gm[col] * (row[i] - mu) * rs + bt[col];
  }
}

// ---------------------------------------------------------------------------
// Weight packer: W[K,128] fp32 row-major -> bf16 WMMA-B fragment layout.
// ---------------------------------------------------------------------------
__global__ __launch_bounds__(256) void pack_w(const float* __restrict__ W,
                                              __bf16* __restrict__ out,
                                              int ktiles) {
  int idx = blockIdx.x * 256 + threadIdx.x;
  int total = ktiles * 4096;
  if (idx >= total) return;
  int h    = idx & 15;
  int lane = (idx >> 4) & 31;
  int t    = (idx >> 9) & 7;
  int kt   = idx >> 12;
  int n  = t * 16 + (lane & 15);
  int hb = lane >> 4;
  int kl = (h < 8) ? (hb * 8 + h) : (16 + hb * 8 + (h - 8));
  int k  = kt * 32 + kl;
  out[idx] = f2bf(W[(size_t)k * HID + n]);
}

__global__ __launch_bounds__(256) void zero_f32(float* __restrict__ p, int n) {
  int i = blockIdx.x * 256 + threadIdx.x;
  if (i < n) p[i] = 0.f;
}

// ---------------------------------------------------------------------------
// Edge-embedding MLP: [E,3] -> silu(.@W1+b1) -> GEMM2+silu+LN -> eOut bf16.
// K=3 first layer done with scalar FMAs (too small for WMMA), second GEMM
// uses the shared WMMA tail.
// ---------------------------------------------------------------------------
__global__ __launch_bounds__(128) void emb_mlp_kernel(
    const float* __restrict__ attr, const float* __restrict__ W1,
    const float* __restrict__ b1, const __bf16* __restrict__ W2p,
    const float* __restrict__ b2, const float* __restrict__ gm,
    const float* __restrict__ bt, __bf16* __restrict__ eOut) {
  __shared__ __bf16 ldsH[4][16 * STRH];
  __shared__ float  ldsF[4][16 * STRF];
  int wave = threadIdx.x >> 5, lane = threadIdx.x & 31;
  int lh = lane & 15, hb = lane >> 4;
  size_t e = ((size_t)blockIdx.x * 4 + wave) * 16 + lh;

  const float* A = attr + e * 3;
  float a0 = A[0], a1 = A[1], a2 = A[2];
  for (int i = 0; i < 64; ++i) {
    int col = hb * 64 + i;
    float h = fmaf(a0, W1[col],
              fmaf(a1, W1[HID + col],
              fmaf(a2, W1[2 * HID + col], b1[col])));
    ldsH[wave][lh * STRH + col] = f2bf(silu(h));
  }
  __syncthreads();
  gemm2_silu_ln(ldsH[wave], ldsF[wave], W2p, b2, gm, bt, lh, hb, lane);
  __bf16* outRow = eOut + e * HID;
  const float* res = ldsF[wave] + lh * STRF;
#pragma unroll
  for (int i = 0; i < 64; ++i) {
    int col = hb * 64 + i;
    outRow[col] = f2bf(res[col]);
  }
}

// ---------------------------------------------------------------------------
// Edge MLP layer: msg = MLP(concat[x[dst], x[src], eIn]) + eIn (residual),
// write eOut bf16, scatter-add fp32 into agg[dst].
// ---------------------------------------------------------------------------
__global__ __launch_bounds__(128) void edge_layer_kernel(
    const float* __restrict__ x, const __bf16* __restrict__ eIn,
    __bf16* __restrict__ eOut, const long long* __restrict__ eidx,
    const __bf16* __restrict__ W1p, const float* __restrict__ b1,
    const __bf16* __restrict__ W2p, const float* __restrict__ b2,
    const float* __restrict__ gm, const float* __restrict__ bt,
    float* __restrict__ agg) {
  __shared__ __bf16 ldsH[4][16 * STRH];
  __shared__ float  ldsF[4][16 * STRF];
  int wave = threadIdx.x >> 5, lane = threadIdx.x & 31;
  int lh = lane & 15, hb = lane >> 4;
  size_t e = ((size_t)blockIdx.x * 4 + wave) * 16 + lh;

  long long srcIdx = eidx[e];            // edge_index[0] = source (x_j)
  long long dstIdx = eidx[N_EDGES + e];  // edge_index[1] = target (x_i)
  const float*  xi = x + (size_t)dstIdx * HID;
  const float*  xj = x + (size_t)srcIdx * HID;
  const __bf16* er = eIn + e * HID;

  // GEMM1: [16,384] @ W1 -> [16,128], K-steps 0-3:x_i, 4-7:x_j, 8-11:edge
  v8f acc[8] = {};
#pragma unroll
  for (int ks = 0; ks < 12; ++ks) {
    v16bf a;
    if (ks < 4)      fill_a_f32(a, xi, ks * 32, hb);
    else if (ks < 8) fill_a_f32(a, xj, (ks - 4) * 32, hb);
    else             fill_a_bf16(a, er, (ks - 8) * 32, hb);
#pragma unroll
    for (int t = 0; t < 8; ++t)
      acc[t] = wmma_bf16(a, loadB(W1p, ks, t, lane), acc[t]);
  }
#pragma unroll
  for (int t = 0; t < 8; ++t) {
    float bias = b1[t * 16 + lh];
#pragma unroll
    for (int g = 0; g < 8; ++g)
      ldsH[wave][(g + hb * 8) * STRH + t * 16 + lh] = f2bf(silu(acc[t][g] + bias));
  }
  __syncthreads();
  gemm2_silu_ln(ldsH[wave], ldsF[wave], W2p, b2, gm, bt, lh, hb, lane);

  __bf16* outRow = eOut + e * HID;
  float*  aggRow = agg + (size_t)dstIdx * HID;
  const float* res = ldsF[wave] + lh * STRF;
#pragma unroll
  for (int i = 0; i < 64; ++i) {
    int col = hb * 64 + i;
    float v = res[col] + bf2f(er[col]);      // residual
    outRow[col] = f2bf(v);
    unsafeAtomicAdd(&aggRow[col], v);        // global_atomic_add_f32, L2-resident
  }
}

// ---------------------------------------------------------------------------
// Node MLP layer: x_next = MLP(concat[x, agg]) + x.
// ---------------------------------------------------------------------------
__global__ __launch_bounds__(128) void node_layer_kernel(
    const float* __restrict__ x, const float* __restrict__ agg,
    const __bf16* __restrict__ W1p, const float* __restrict__ b1,
    const __bf16* __restrict__ W2p, const float* __restrict__ b2,
    const float* __restrict__ gm, const float* __restrict__ bt,
    float* __restrict__ xNext, int tiles) {
  __shared__ __bf16 ldsH[4][16 * STRH];
  __shared__ float  ldsF[4][16 * STRF];
  int wave = threadIdx.x >> 5, lane = threadIdx.x & 31;
  int lh = lane & 15, hb = lane >> 4;
  int tile = blockIdx.x * 4 + wave;
  bool active = tile < tiles;
  if (!active) tile = 0;                    // keep control flow uniform
  size_t row = (size_t)tile * 16 + lh;
  const float* xr = x + row * HID;
  const float* ar = agg + row * HID;

  v8f acc[8] = {};
#pragma unroll
  for (int ks = 0; ks < 8; ++ks) {          // K=256: 0-3 x, 4-7 agg
    v16bf a;
    if (ks < 4) fill_a_f32(a, xr, ks * 32, hb);
    else        fill_a_f32(a, ar, (ks - 4) * 32, hb);
#pragma unroll
    for (int t = 0; t < 8; ++t)
      acc[t] = wmma_bf16(a, loadB(W1p, ks, t, lane), acc[t]);
  }
#pragma unroll
  for (int t = 0; t < 8; ++t) {
    float bias = b1[t * 16 + lh];
#pragma unroll
    for (int g = 0; g < 8; ++g)
      ldsH[wave][(g + hb * 8) * STRH + t * 16 + lh] = f2bf(silu(acc[t][g] + bias));
  }
  __syncthreads();
  gemm2_silu_ln(ldsH[wave], ldsF[wave], W2p, b2, gm, bt, lh, hb, lane);

  if (active) {
    float* outRow = xNext + row * HID;
    const float* res = ldsF[wave] + lh * STRF;
#pragma unroll
    for (int i = 0; i < 64; ++i) {
      int col = hb * 64 + i;
      outRow[col] = res[col] + xr[col];     // residual
    }
  }
}

// ---------------------------------------------------------------------------
// Host orchestration.
// ---------------------------------------------------------------------------
extern "C" void kernel_launch(void* const* d_in, const int* in_sizes, int n_in,
                              void* d_out, int out_size, void* d_ws, size_t ws_size,
                              hipStream_t stream) {
  (void)in_sizes; (void)n_in; (void)out_size; (void)ws_size;
  const float* x         = (const float*)d_in[0];
  const float* edge_attr = (const float*)d_in[1];
  const float* emb_W1    = (const float*)d_in[2];
  const float* emb_b1    = (const float*)d_in[3];
  const float* emb_W2    = (const float*)d_in[4];
  const float* emb_b2    = (const float*)d_in[5];
  const float* emb_g     = (const float*)d_in[6];
  const float* emb_beta  = (const float*)d_in[7];
  const float* eW1       = (const float*)d_in[8];
  const float* eb1       = (const float*)d_in[9];
  const float* eW2       = (const float*)d_in[10];
  const float* eb2       = (const float*)d_in[11];
  const float* eg        = (const float*)d_in[12];
  const float* ebeta     = (const float*)d_in[13];
  const float* nW1       = (const float*)d_in[14];
  const float* nb1       = (const float*)d_in[15];
  const float* nW2       = (const float*)d_in[16];
  const float* nb2       = (const float*)d_in[17];
  const float* ng        = (const float*)d_in[18];
  const float* nbeta     = (const float*)d_in[19];
  const long long* eidx  = (const long long*)d_in[20];

  char* w = (char*)d_ws;
  size_t off = 0;
  auto carve = [&](size_t bytes) -> char* {
    char* p = w + off;
    off = (off + bytes + 255) & ~(size_t)255;
    return p;
  };
  __bf16* eA0 = (__bf16*)carve((size_t)N_EDGES * HID * 2);  // 131 MB
  __bf16* eA1 = (__bf16*)carve((size_t)N_EDGES * HID * 2);  // 131 MB
  float*  agg = (float*)carve((size_t)N_NODES * HID * 4);   // 25.6 MB
  float*  x1  = (float*)carve((size_t)N_NODES * HID * 4);   // 25.6 MB
  __bf16* pEmbW2 = (__bf16*)carve((size_t)4 * 4096 * 2);
  __bf16 *pEW1[2], *pEW2[2], *pNW1[2], *pNW2[2];
  for (int l = 0; l < 2; ++l) {
    pEW1[l] = (__bf16*)carve((size_t)12 * 4096 * 2);
    pEW2[l] = (__bf16*)carve((size_t)4 * 4096 * 2);
    pNW1[l] = (__bf16*)carve((size_t)8 * 4096 * 2);
    pNW2[l] = (__bf16*)carve((size_t)4 * 4096 * 2);
  }

  // Pack weights into WMMA-B fragment layout (bf16). Tiny, runs every call.
  auto pack = [&](const float* src, __bf16* dst, int ktiles) {
    int total = ktiles * 4096;
    pack_w<<<(total + 255) / 256, 256, 0, stream>>>(src, dst, ktiles);
  };
  pack(emb_W2, pEmbW2, 4);
  for (int l = 0; l < 2; ++l) {
    pack(eW1 + (size_t)l * 384 * HID, pEW1[l], 12);
    pack(eW2 + (size_t)l * HID * HID, pEW2[l], 4);
    pack(nW1 + (size_t)l * 256 * HID, pNW1[l], 8);
    pack(nW2 + (size_t)l * HID * HID, pNW2[l], 4);
  }

  // Edge embedding MLP -> eA0 (bf16)
  emb_mlp_kernel<<<N_EDGES / 64, 128, 0, stream>>>(
      edge_attr, emb_W1, emb_b1, pEmbW2, emb_b2, emb_g, emb_beta, eA0);

  const float* xcur = x;
  __bf16 *ein = eA0, *eout = eA1;
  for (int l = 0; l < 2; ++l) {
    zero_f32<<<(N_NODES * HID + 255) / 256, 256, 0, stream>>>(agg, N_NODES * HID);
    edge_layer_kernel<<<N_EDGES / 64, 128, 0, stream>>>(
        xcur, ein, eout, eidx,
        pEW1[l], eb1 + l * HID, pEW2[l], eb2 + l * HID,
        eg + l * HID, ebeta + l * HID, agg);
    float* xnext = (l == 0) ? x1 : (float*)d_out;
    int tiles = N_NODES / 16;  // 3125, exact
    node_layer_kernel<<<(tiles + 3) / 4, 128, 0, stream>>>(
        xcur, agg,
        pNW1[l], nb1 + l * HID, pNW2[l], nb2 + l * HID,
        ng + l * HID, nbeta + l * HID, xnext, tiles);
    xcur = xnext;
    __bf16* t = ein; ein = eout; eout = t;
  }
}
